// GNN_CL_66486093742318
// MI455X (gfx1250) — compile-verified, matrix operand order checked by reading, not run on hardware
//
#include <hip/hip_runtime.h>
#include <hip/hip_bf16.h>

typedef __attribute__((ext_vector_type(16))) _Float16 v16h;
typedef __attribute__((ext_vector_type(8)))  float    v8f;

#define NNODES 50000
#define NEDGES 800000
#define ETOT   850000      // NEDGES + NNODES self loops
#define MT     3125        // 50000 / 16 row tiles (exact)

// ---------------- helpers ----------------
static __device__ __forceinline__ unsigned fkey(float x) {
  unsigned b = __float_as_uint(x);
  return (b & 0x80000000u) ? ~b : (b | 0x80000000u);
}
static __device__ __forceinline__ float funkey(unsigned k) {
  return (k & 0x80000000u) ? __uint_as_float(k ^ 0x80000000u)
                           : __uint_as_float(~k);
}

__global__ void fill_u32(unsigned* __restrict__ p, unsigned v, int n) {
  int t = blockIdx.x * blockDim.x + threadIdx.x;
  if (t < n) p[t] = v;
}

// ---- pack A [50000 x K] f32 row-major -> WMMA 16x32 f16 A-fragments ----
// Fragment layout (ISA 7.12.2, 16-bit A 16x32): lane = M%16 (+16 for K-half),
// dword i<4 holds K = 2i,2i+1 (+8*khalf); i>=4 holds K = 16+2(i-4) (+8*khalf)
__global__ void pack_a(const float* __restrict__ A, _Float16* __restrict__ out,
                       int K, int KS) {
  int t = blockIdx.x * blockDim.x + threadIdx.x;
  if (t >= MT * KS * 32) return;
  int lane = t & 31;
  int ks   = (t >> 5) % KS;
  int mt   = t / (32 * KS);
  int m    = mt * 16 + (lane & 15);
  int kh   = lane >> 4;
  const float* row = A + (size_t)m * K + ks * 32;
  v16h f;
#pragma unroll
  for (int i = 0; i < 8; ++i) {
    int k = (i < 4) ? (2 * i + 8 * kh) : (16 + 2 * (i - 4) + 8 * kh);
    f[2 * i]     = (_Float16)row[k];
    f[2 * i + 1] = (_Float16)row[k + 1];
  }
  *(v16h*)(out + (size_t)t * 16) = f;
}

// ---- pack B [K x Nout] f32 row-major -> WMMA 32x16 f16 B-fragments ----
__global__ void pack_b(const float* __restrict__ B, _Float16* __restrict__ out,
                       int K, int Nout, int NT, int KS) {
  int t = blockIdx.x * blockDim.x + threadIdx.x;
  if (t >= NT * KS * 32) return;
  int lane = t & 31;
  int ks   = (t >> 5) % KS;
  int nt   = t / (32 * KS);
  int n    = nt * 16 + (lane & 15);
  int kh   = lane >> 4;
  v16h f;
#pragma unroll
  for (int i = 0; i < 8; ++i) {
    int k = ks * 32 + ((i < 4) ? (2 * i + 8 * kh) : (16 + 2 * (i - 4) + 8 * kh));
    f[2 * i]     = (_Float16)B[(size_t)k * Nout + n];
    f[2 * i + 1] = (_Float16)B[(size_t)(k + 1) * Nout + n];
  }
  *(v16h*)(out + (size_t)t * 16) = f;
}

// ---- WMMA GEMM: one wave computes one 16x16 f32 tile; 8 waves / block ----
// EPI 0: plain store. EPI 1: += bias[col], sigmoid, store (final layer).
template <int KS, int EPI>
__global__ void gemm_wmma(const _Float16* __restrict__ Ap,
                          const _Float16* __restrict__ Bp,
                          float* __restrict__ C, int NT, int Nout,
                          const float* __restrict__ bias) {
  int lane = threadIdx.x & 31;
  int wave = threadIdx.x >> 5;
  int nt   = blockIdx.y * 8 + wave;
  if (nt >= NT) return;                 // wave-uniform: EXEC stays all-ones
  int mt = blockIdx.x;

  const v16h* Av = (const v16h*)Ap + (size_t)(mt * KS) * 32 + lane;
  const v16h* Bv = (const v16h*)Bp + (size_t)(nt * KS) * 32 + lane;
  v8f acc = {};
#pragma unroll
  for (int ks = 0; ks < KS; ++ks) {
    v16h a = Av[ks * 32];
    v16h b = Bv[ks * 32];
    acc = __builtin_amdgcn_wmma_f32_16x16x32_f16(
        false, a, false, b, (short)0, acc, false, false);
  }
  int col   = nt * 16 + (lane & 15);
  int rbase = (lane >> 4) * 8;          // lanes 16-31 hold rows M=8..15
  float* outp = C + (size_t)(mt * 16) * Nout + col;
  if (EPI == 0) {
#pragma unroll
    for (int r = 0; r < 8; ++r)
      outp[(size_t)(rbase + r) * Nout] = acc[r];
  } else {
    float bb = bias[col];
#pragma unroll
    for (int r = 0; r < 8; ++r) {
      float v = acc[r] + bb;
      outp[(size_t)(rbase + r) * Nout] = 1.0f / (1.0f + __expf(-v));
    }
  }
}

// ---- per-node attention dot products: a_s[n,h], a_d[n,h] ----
__global__ void node_attn(const float* __restrict__ h,
                          const float* __restrict__ asrc,
                          const float* __restrict__ adst,
                          float* __restrict__ as_, float* __restrict__ ad_,
                          int H, int C) {
  int t = blockIdx.x * blockDim.x + threadIdx.x;
  if (t >= NNODES * H) return;
  int n = t / H, hh = t % H;
  const float4* hp = (const float4*)(h + ((size_t)n * H + hh) * C);
  const float4* sp = (const float4*)(asrc + (size_t)hh * C);
  const float4* dp = (const float4*)(adst + (size_t)hh * C);
  float s = 0.f, d = 0.f;
#pragma unroll 4
  for (int i = 0; i < C / 4; ++i) {
    float4 hv = hp[i], sv = sp[i], dv = dp[i];
    s += hv.x * sv.x + hv.y * sv.y + hv.z * sv.z + hv.w * sv.w;
    d += hv.x * dv.x + hv.y * dv.y + hv.z * dv.z + hv.w * dv.w;
  }
  as_[t] = s;
  ad_[t] = d;
}

// ---- edge pass 1: leaky_relu(a_s[src]+a_d[dst]), segment max via u32 keys ----
__global__ void edge_score_max(const int* __restrict__ ei,
                               const float* __restrict__ as_,
                               const float* __restrict__ ad_,
                               float* __restrict__ ebuf,
                               unsigned* __restrict__ mkey, int H) {
  int t = blockIdx.x * blockDim.x + threadIdx.x;
  if (t >= ETOT) return;
  int s, d;
  if (t < NEDGES) { s = ei[t]; d = ei[NEDGES + t]; }
  else            { s = d = t - NEDGES; }
  for (int hh = 0; hh < H; ++hh) {
    float v = as_[(size_t)s * H + hh] + ad_[(size_t)d * H + hh];
    v = v > 0.f ? v : 0.2f * v;                 // leaky_relu(0.2)
    ebuf[(size_t)t * H + hh] = v;
    atomicMax(&mkey[(size_t)d * H + hh], fkey(v));
  }
}

// ---- edge pass 2: w = exp(e - m[dst]); scatter-add denom + w*h[src] ----
__global__ void edge_accum(const int* __restrict__ ei,
                           const float* __restrict__ ebuf,
                           const unsigned* __restrict__ mkey,
                           const float* __restrict__ h,
                           float* __restrict__ denom, float* __restrict__ agg,
                           int H, int C) {
  int t = blockIdx.x * blockDim.x + threadIdx.x;
  if (t >= ETOT) return;
  int s, d;
  if (t < NEDGES) { s = ei[t]; d = ei[NEDGES + t]; }
  else            { s = d = t - NEDGES; }
  for (int hh = 0; hh < H; ++hh) {
    float m = funkey(mkey[(size_t)d * H + hh]);
    float w = __expf(ebuf[(size_t)t * H + hh] - m);
    atomicAdd(&denom[(size_t)d * H + hh], w);
    const float4* hp = (const float4*)(h + ((size_t)s * H + hh) * C);
    float* ap = agg + ((size_t)d * H + hh) * C;
    for (int i = 0; i < C / 4; ++i) {
      float4 hv = hp[i];
      atomicAdd(ap + 4 * i + 0, hv.x * w);
      atomicAdd(ap + 4 * i + 1, hv.y * w);
      atomicAdd(ap + 4 * i + 2, hv.z * w);
      atomicAdd(ap + 4 * i + 3, hv.w * w);
    }
  }
}

// ---- finalize: out = elu(agg / (denom + eps) + bias)  (concat layout) ----
__global__ void node_finalize(const float* __restrict__ agg,
                              const float* __restrict__ denom,
                              const float* __restrict__ bias,
                              float* __restrict__ out, int H, int C) {
  int D = H * C;
  int t = blockIdx.x * blockDim.x + threadIdx.x;
  if (t >= NNODES * D) return;
  int n = t / D, j = t % D, hh = j / C;
  float v = agg[t] / (denom[(size_t)n * H + hh] + 1e-16f) + bias[j];
  out[t] = v > 0.f ? v : (__expf(v) - 1.0f);
}

// ---------------- host ----------------
extern "C" void kernel_launch(void* const* d_in, const int* in_sizes, int n_in,
                              void* d_out, int out_size, void* d_ws, size_t ws_size,
                              hipStream_t stream) {
  const float* x      = (const float*)d_in[0];
  const int*   ei     = (const int*)d_in[1];     // edge_index [2,E]
  const float* W1     = (const float*)d_in[2];   // [128,128]
  const float* asrc1  = (const float*)d_in[3];   // [2,64]
  const float* adst1  = (const float*)d_in[4];
  const float* b1     = (const float*)d_in[5];   // [128]
  const float* W2     = (const float*)d_in[6];   // [128,64]
  const float* asrc2  = (const float*)d_in[7];   // [1,64]
  const float* adst2  = (const float*)d_in[8];
  const float* b2     = (const float*)d_in[9];   // [64]
  const float* linw   = (const float*)d_in[10];  // [64,16]
  const float* linb   = (const float*)d_in[11];  // [16]
  float* out = (float*)d_out;
  char* base = (char*)d_ws;

  // workspace layout (manually reused, ~94 MiB total)
  _Float16* Bpack1 = (_Float16*)(base + 0);          // 8*4*512*16b  = 32 KB
  _Float16* Bpack2 = (_Float16*)(base + 32768);      // 4*4*512*16b  = 16 KB
  _Float16* Bpack3 = (_Float16*)(base + 49152);      // 1*2*512*16b  =  2 KB
  float*    a_s1   = (float*)(base + 51200);         // N*2
  float*    a_d1   = (float*)(base + 451200);
  float*    denom1 = (float*)(base + 851200);
  unsigned* mkey1  = (unsigned*)(base + 1251200);
  float*    a_s2   = (float*)(base + 1651200);       // N*1
  float*    a_d2   = (float*)(base + 1851200);
  float*    denom2 = (float*)(base + 2051200);
  unsigned* mkey2  = (unsigned*)(base + 2251200);
  float*    ebuf   = (float*)(base + 2451200);       // ET*2 f32 (reused L2)
  _Float16* Apack  = (_Float16*)(base + 9251200);    // 3125*4*512*16b (reused)
  float*    hraw   = (float*)(base + 22051200);      // N*128 f32 (L1 raw)
  float*    h2     = (float*)(base + 34851200);      // N*64  f32 (aliases hraw tail)
  float*    h1     = (float*)(base + 47651200);      // N*128 f32
  float*    agg    = (float*)(base + 73251200);      // N*128 f32
  (void)n_in; (void)in_sizes; (void)out_size; (void)ws_size;

  const int TB = 256;
  auto nb = [](int n) { return (n + 255) / 256; };

  // ================= Layer 1: GATConv(128 -> 2 heads x 64, concat) =========
  pack_b<<<nb(8 * 4 * 32), TB, 0, stream>>>(W1, Bpack1, 128, 128, 8, 4);
  pack_a<<<nb(MT * 4 * 32), TB, 0, stream>>>(x, Apack, 128, 4);
  gemm_wmma<4, 0><<<dim3(MT, 1), TB, 0, stream>>>(Apack, Bpack1, hraw, 8, 128, nullptr);

  node_attn<<<nb(NNODES * 2), TB, 0, stream>>>(hraw, asrc1, adst1, a_s1, a_d1, 2, 64);
  fill_u32<<<nb(NNODES * 2), TB, 0, stream>>>(mkey1, 0u, NNODES * 2);
  fill_u32<<<nb(NNODES * 2), TB, 0, stream>>>((unsigned*)denom1, 0u, NNODES * 2);
  fill_u32<<<nb(NNODES * 128), TB, 0, stream>>>((unsigned*)agg, 0u, NNODES * 128);
  edge_score_max<<<nb(ETOT), TB, 0, stream>>>(ei, a_s1, a_d1, ebuf, mkey1, 2);
  edge_accum<<<nb(ETOT), TB, 0, stream>>>(ei, ebuf, mkey1, hraw, denom1, agg, 2, 64);
  node_finalize<<<nb(NNODES * 128), TB, 0, stream>>>(agg, denom1, b1, h1, 2, 64);

  // ================= Layer 2: GATConv(128 -> 1 head x 64, mean) ============
  pack_b<<<nb(4 * 4 * 32), TB, 0, stream>>>(W2, Bpack2, 128, 64, 4, 4);
  pack_a<<<nb(MT * 4 * 32), TB, 0, stream>>>(h1, Apack, 128, 4);
  gemm_wmma<4, 0><<<dim3(MT, 1), TB, 0, stream>>>(Apack, Bpack2, hraw, 4, 64, nullptr);

  node_attn<<<nb(NNODES), TB, 0, stream>>>(hraw, asrc2, adst2, a_s2, a_d2, 1, 64);
  fill_u32<<<nb(NNODES), TB, 0, stream>>>(mkey2, 0u, NNODES);
  fill_u32<<<nb(NNODES), TB, 0, stream>>>((unsigned*)denom2, 0u, NNODES);
  fill_u32<<<nb(NNODES * 64), TB, 0, stream>>>((unsigned*)agg, 0u, NNODES * 64);
  edge_score_max<<<nb(ETOT), TB, 0, stream>>>(ei, a_s2, a_d2, ebuf, mkey2, 1);
  edge_accum<<<nb(ETOT), TB, 0, stream>>>(ei, ebuf, mkey2, hraw, denom2, agg, 1, 64);
  node_finalize<<<nb(NNODES * 64), TB, 0, stream>>>(agg, denom2, b2, h2, 1, 64);

  // ================= Final: sigmoid(h2 @ lin_w + lin_b) into d_out =========
  pack_b<<<nb(1 * 2 * 32), TB, 0, stream>>>(linw, Bpack3, 64, 16, 1, 2);
  pack_a<<<nb(MT * 2 * 32), TB, 0, stream>>>(h2, Apack, 64, 2);
  gemm_wmma<2, 1><<<dim3(MT, 1), TB, 0, stream>>>(Apack, Bpack3, out, 1, 16, linb);
}
